// DynamicGaussianModel_7791070675875
// MI455X (gfx1250) — compile-verified, matrix operand order checked by reading, not run on hardware
//
#include <hip/hip_runtime.h>
#include <hip/hip_bf16.h>

typedef __attribute__((ext_vector_type(16))) _Float16 v16h;
typedef __attribute__((ext_vector_type(8)))  _Float16 v8h;
typedef __attribute__((ext_vector_type(8)))  float    v8f;

#define ASTRIDE 320          // activation tile row stride (halfs)
#define TILE_M  128          // rows per workgroup
#define LDS_AT_BYTES (TILE_M * ASTRIDE * 2)     // 81920
#define LDS_WB_BYTES (320 * 256 * 2)            // 163840 (largest layer: W2)
#define LDS_HW_BYTES (256 * 12 * 4)             // 12288
#define LDS_TOTAL    (LDS_AT_BYTES + LDS_WB_BYTES + LDS_HW_BYTES)   // 258048

// half-element offsets inside d_ws
#define W0OFF 0              // [256][64]
#define W1OFF 16384          // [256][256]
#define W2OFF 81920          // [256][320]
#define W3OFF 163840         // [256][256]
#define HEAD_BYTE_OFF ((163840 + 65536) * 2)    // f32 [256][12]

// ---------------------------------------------------------------------------
// Prep: transpose & convert weights to f16 N-major, pack head weights.
// grid=(256 blocks: n), block=(256 threads: k)
// ---------------------------------------------------------------------------
__global__ void dgm_prep(const float* __restrict__ W0, const float* __restrict__ W1,
                         const float* __restrict__ W2, const float* __restrict__ W3,
                         const float* __restrict__ Wx, const float* __restrict__ Wr,
                         const float* __restrict__ Ws, const float* __restrict__ Wo,
                         _Float16* __restrict__ wsH, float* __restrict__ hwg) {
    int n = blockIdx.x;      // output column 0..255
    int k = threadIdx.x;     // input row 0..255
    if (k < 64)
        wsH[W0OFF + n * 64 + k] = (k < 60) ? (_Float16)W0[k * 256 + n] : (_Float16)0.0f;
    wsH[W1OFF + n * 256 + k] = (_Float16)W1[k * 256 + n];
    wsH[W2OFF + n * 320 + k] = (_Float16)W2[k * 256 + n];
    if (k < 64) {
        int k2 = 256 + k;
        wsH[W2OFF + n * 320 + k2] = (k2 < 316) ? (_Float16)W2[k2 * 256 + n] : (_Float16)0.0f;
    }
    wsH[W3OFF + n * 256 + k] = (_Float16)W3[k * 256 + n];
    if (n == 0) {
        float* h = hwg + k * 12;
        h[0]  = Wx[k * 3 + 0]; h[1] = Wx[k * 3 + 1]; h[2] = Wx[k * 3 + 2];
        h[3]  = Wr[k * 4 + 0]; h[4] = Wr[k * 4 + 1]; h[5] = Wr[k * 4 + 2]; h[6] = Wr[k * 4 + 3];
        h[7]  = Ws[k * 3 + 0]; h[8] = Ws[k * 3 + 1]; h[9] = Ws[k * 3 + 2];
        h[10] = Wo[k];         h[11] = 0.0f;
    }
}

// ---------------------------------------------------------------------------
// 16-bit WMMA fragment load: two contiguous 8-half chunks (k+off, k+off+16).
// Works for both A (row = M = lane&15) and B (row = N = lane&15) with the
// N-major weight layout.
// ---------------------------------------------------------------------------
__device__ __forceinline__ v16h ld_frag(const _Float16* p) {
    v8h lo = *(const v8h*)p;
    v8h hi = *(const v8h*)(p + 16);
    v16h r;
#pragma unroll
    for (int i = 0; i < 8; ++i) { r[i] = lo[i]; r[i + 8] = hi[i]; }
    return r;
}

// ---------------------------------------------------------------------------
// Stage one layer's N-major f16 weights (contiguous in ws) into LDS.
// Flat coalesced copy in 16-byte chunks by all 256 threads.
// ---------------------------------------------------------------------------
__device__ __forceinline__ void stage_wbuf(_Float16* __restrict__ WB,
                                           const _Float16* __restrict__ WT,
                                           int halfs, int tid) {
    const v8h* __restrict__ src = (const v8h*)WT;
    v8h* __restrict__ dst = (v8h*)WB;
    int n16 = halfs >> 3;
    for (int i = tid; i < n16; i += 256) dst[i] = src[i];
}

// ---------------------------------------------------------------------------
// One MLP layer: h_out[128x256] = relu(A[128xK] @ W[Kx256] + b)
// Per wave: 16 rows, full 256 cols as 16 accumulator tiles.
// B fragments come from the LDS-staged weight buffer (shared by all 8 waves).
// ---------------------------------------------------------------------------
template <int KCH>   // K = KCH*32
__device__ __forceinline__ void mlp_layer(_Float16* AT, const _Float16* WB,
                                          int rowbase, int lane, int acol,
                                          const float* __restrict__ bias) {
    const int K    = KCH * 32;
    const int lrow = lane & 15;
    const int off  = (lane >> 4) << 3;   // 0 or 8

    v8f acc[16];
#pragma unroll
    for (int nt = 0; nt < 16; ++nt)
#pragma unroll
        for (int v = 0; v < 8; ++v) acc[nt][v] = 0.0f;

    const _Float16* arow = AT + (rowbase + lrow) * ASTRIDE + acol + off;
    const _Float16* brow = WB + lrow * K + off;
    for (int kc = 0; kc < K; kc += 32) {
        v16h a = ld_frag(arow + kc);
#pragma unroll
        for (int nt = 0; nt < 16; ++nt) {
            v16h b = ld_frag(brow + nt * 16 * K + kc);
            acc[nt] = __builtin_amdgcn_wmma_f32_16x16x32_f16(
                false, a, false, b, (short)0, acc[nt], false, false);
        }
    }

    // bias + relu + f16 writeback (D layout: row = v + 8*(lane>=16), col = nt*16 + lane&15)
    const int r0 = rowbase + ((lane >> 4) << 3);
#pragma unroll
    for (int nt = 0; nt < 16; ++nt) {
        int col  = nt * 16 + lrow;
        float bv = bias[col];
#pragma unroll
        for (int v = 0; v < 8; ++v) {
            float x = fmaxf(acc[nt][v] + bv, 0.0f);
            AT[(r0 + v) * ASTRIDE + col] = (_Float16)x;
        }
    }
}

// ---------------------------------------------------------------------------
// Main fused kernel
// ---------------------------------------------------------------------------
__global__ __launch_bounds__(256, 1) void dgm_main(
    const float* __restrict__ xyz,  const float* __restrict__ rot,
    const float* __restrict__ scal, const float* __restrict__ opac,
    const float* __restrict__ timep,
    const float* __restrict__ b0, const float* __restrict__ b1,
    const float* __restrict__ b2, const float* __restrict__ b3,
    const float* __restrict__ bx, const float* __restrict__ br,
    const float* __restrict__ bsc, const float* __restrict__ bo,
    const _Float16* __restrict__ wsH, const float* __restrict__ hwg,
    float* __restrict__ out, int npts) {
    extern __shared__ char smem[];
    _Float16* AT    = (_Float16*)smem;
    _Float16* WB    = (_Float16*)(smem + LDS_AT_BYTES);
    float*    HWlds = (float*)(smem + LDS_AT_BYTES + LDS_WB_BYTES);

    const int tid     = threadIdx.x;
    const int lane    = tid & 31;
    const int wave    = tid >> 5;
    const int rowbase = wave * 16;
    const int pbase   = blockIdx.x * TILE_M;

    // stage head weights [256][12] f32 into LDS
    {
        const float* g = hwg + tid * 12;
        float*       l = HWlds + tid * 12;
#pragma unroll
        for (int j = 0; j < 12; ++j) l[j] = g[j];
    }

    const float PI = 3.14159265358979323846f;

    // ---- positional + time encoding -> ATILE cols 256..319 ----
    if (lane < 16) {
        int p  = pbase + rowbase + lane;
        int pc = p < npts ? p : npts - 1;
        float c3[3] = { xyz[pc * 3 + 0], xyz[pc * 3 + 1], xyz[pc * 3 + 2] };
        float t = timep[0];
        _Float16* arow = AT + (rowbase + lane) * ASTRIDE + 256;
        arow[0] = (_Float16)c3[0]; arow[1] = (_Float16)c3[1]; arow[2] = (_Float16)c3[2];
#pragma unroll
        for (int c = 0; c < 3; ++c) {
            float f = PI;
#pragma unroll
            for (int fi = 0; fi < 6; ++fi) {
                float sv = c3[c] * f;
                arow[3 + c * 12 + fi]     = (_Float16)__sinf(sv);
                arow[3 + c * 12 + 6 + fi] = (_Float16)__cosf(sv);
                f *= 2.0f;
            }
        }
        arow[39] = (_Float16)t;
        {
            float f = PI;
#pragma unroll
            for (int fi = 0; fi < 10; ++fi) {
                float sv = t * f;
                arow[40 + fi] = (_Float16)__sinf(sv);
                arow[50 + fi] = (_Float16)__cosf(sv);
                f *= 2.0f;
            }
        }
        arow[60] = (_Float16)0.0f; arow[61] = (_Float16)0.0f;
        arow[62] = (_Float16)0.0f; arow[63] = (_Float16)0.0f;
    }

    // ---- 4 GEMM layers (WMMA), weights staged whole-layer into LDS ----
    stage_wbuf(WB, wsH + W0OFF, 256 * 64, tid);
    __syncthreads();
    mlp_layer<2 >(AT, WB, rowbase, lane, 256, b0);   // K=64  (x_input)
    __syncthreads();

    stage_wbuf(WB, wsH + W1OFF, 256 * 256, tid);
    __syncthreads();
    mlp_layer<8 >(AT, WB, rowbase, lane, 0, b1);     // K=256
    __syncthreads();

    stage_wbuf(WB, wsH + W2OFF, 256 * 320, tid);
    __syncthreads();
    mlp_layer<10>(AT, WB, rowbase, lane, 0, b2);     // K=320 (h ++ x_input skip)
    __syncthreads();

    stage_wbuf(WB, wsH + W3OFF, 256 * 256, tid);
    __syncthreads();
    mlp_layer<8 >(AT, WB, rowbase, lane, 0, b3);     // K=256

    // ---- heads + quaternion/covariance epilogue ----
    if (lane < 16) {
        int p = pbase + rowbase + lane;
        const _Float16* arow = AT + (rowbase + lane) * ASTRIDE;
        float s[11];
#pragma unroll
        for (int j = 0; j < 11; ++j) s[j] = 0.0f;
        for (int k = 0; k < 256; ++k) {
            float av = (float)arow[k];
            const float* hw = HWlds + k * 12;
#pragma unroll
            for (int j = 0; j < 11; ++j) s[j] = fmaf(av, hw[j], s[j]);
        }
        if (p < npts) {
            // xyz out
            float ox = xyz[p * 3 + 0] + s[0] + bx[0];
            float oy = xyz[p * 3 + 1] + s[1] + bx[1];
            float oz = xyz[p * 3 + 2] + s[2] + bx[2];
            // quaternion
            float qw = rot[p * 4 + 0] + s[3] + br[0];
            float qx = rot[p * 4 + 1] + s[4] + br[1];
            float qy = rot[p * 4 + 2] + s[5] + br[2];
            float qz = rot[p * 4 + 3] + s[6] + br[3];
            float nrm = fmaxf(sqrtf(qw * qw + qx * qx + qy * qy + qz * qz), 1e-12f);
            float inv = 1.0f / nrm;
            qw *= inv; qx *= inv; qy *= inv; qz *= inv;
            float r00 = 1.0f - 2.0f * (qy * qy + qz * qz);
            float r01 = 2.0f * (qx * qy - qw * qz);
            float r02 = 2.0f * (qx * qz + qw * qy);
            float r10 = 2.0f * (qx * qy + qw * qz);
            float r11 = 1.0f - 2.0f * (qx * qx + qz * qz);
            float r12 = 2.0f * (qy * qz - qw * qx);
            float r20 = 2.0f * (qx * qz - qw * qy);
            float r21 = 2.0f * (qy * qz + qw * qx);
            float r22 = 1.0f - 2.0f * (qx * qx + qy * qy);
            float s0 = __expf(scal[p * 3 + 0] + s[7] + bsc[0]);
            float s1 = __expf(scal[p * 3 + 1] + s[8] + bsc[1]);
            float s2 = __expf(scal[p * 3 + 2] + s[9] + bsc[2]);
            float m00 = r00 * s0, m01 = r01 * s1, m02 = r02 * s2;
            float m10 = r10 * s0, m11 = r11 * s1, m12 = r12 * s2;
            float m20 = r20 * s0, m21 = r21 * s1, m22 = r22 * s2;
            float c00 = m00 * m00 + m01 * m01 + m02 * m02;
            float c01 = m00 * m10 + m01 * m11 + m02 * m12;
            float c02 = m00 * m20 + m01 * m21 + m02 * m22;
            float c11 = m10 * m10 + m11 * m11 + m12 * m12;
            float c12 = m10 * m20 + m11 * m21 + m12 * m22;
            float c22 = m20 * m20 + m21 * m21 + m22 * m22;
            float ov = opac[p] + s[10] + bo[0];
            float op = 1.0f / (1.0f + __expf(-ov));
            float* o = out + p * 13;
            o[0] = ox;  o[1] = oy;  o[2] = oz;
            o[3] = c00; o[4] = c01; o[5] = c02;
            o[6] = c01; o[7] = c11; o[8] = c12;
            o[9] = c02; o[10] = c12; o[11] = c22;
            o[12] = op;
        }
    }
}

// ---------------------------------------------------------------------------
extern "C" void kernel_launch(void* const* d_in, const int* in_sizes, int n_in,
                              void* d_out, int out_size, void* d_ws, size_t ws_size,
                              hipStream_t stream) {
    const float* xyz   = (const float*)d_in[0];
    const float* rotp  = (const float*)d_in[1];
    const float* scal  = (const float*)d_in[2];
    const float* opac  = (const float*)d_in[3];
    const float* timep = (const float*)d_in[4];
    const float* W0 = (const float*)d_in[5];  const float* b0 = (const float*)d_in[6];
    const float* W1 = (const float*)d_in[7];  const float* b1 = (const float*)d_in[8];
    const float* W2 = (const float*)d_in[9];  const float* b2 = (const float*)d_in[10];
    const float* W3 = (const float*)d_in[11]; const float* b3 = (const float*)d_in[12];
    const float* Wx = (const float*)d_in[13]; const float* bx = (const float*)d_in[14];
    const float* Wr = (const float*)d_in[15]; const float* br = (const float*)d_in[16];
    const float* Ws = (const float*)d_in[17]; const float* bs = (const float*)d_in[18];
    const float* Wo = (const float*)d_in[19]; const float* bo = (const float*)d_in[20];

    _Float16* wsH = (_Float16*)d_ws;
    float*    hwg = (float*)((char*)d_ws + HEAD_BYTE_OFF);
    float*    out = (float*)d_out;
    int npts = in_sizes[0] / 3;

    dgm_prep<<<256, 256, 0, stream>>>(W0, W1, W2, W3, Wx, Wr, Ws, Wo, wsH, hwg);

    (void)hipFuncSetAttribute((const void*)dgm_main,
                              hipFuncAttributeMaxDynamicSharedMemorySize, LDS_TOTAL);
    int nblocks = (npts + TILE_M - 1) / TILE_M;
    dgm_main<<<nblocks, 256, LDS_TOTAL, stream>>>(
        xyz, rotp, scal, opac, timep, b0, b1, b2, b3, bx, br, bs, bo,
        wsH, hwg, out, npts);
}
